// StaticW8A8Int8Linear_61022895341836
// MI455X (gfx1250) — 8-bit matrix operand order repaired and checked by reading, compiled, not run on hardware
//
#include <hip/hip_runtime.h>
#include <hip/hip_bf16.h>

typedef __attribute__((ext_vector_type(8))) int   v8i;
typedef __attribute__((ext_vector_type(4))) int   v4i;
typedef __attribute__((ext_vector_type(2))) int   v2i;
typedef __attribute__((ext_vector_type(4))) float v4f;

#define MDIM 8192
#define KDIM 4096
#define NDIM 4096

__global__ void quantize_x_kernel(const float* __restrict__ x,
                                  const float* __restrict__ input_scale,
                                  signed char* __restrict__ aq) {
    const float inv = 1.0f / input_scale[0];
    size_t i = (size_t)blockIdx.x * blockDim.x + threadIdx.x;
    const float* src = x + i * 16;
    int packed[4];
#pragma unroll
    for (int c = 0; c < 4; ++c) {
        v4f f = *(const v4f*)(src + c * 4);
        int r = 0;
#pragma unroll
        for (int j = 0; j < 4; ++j) {
            float q = __builtin_rintf(f[j] * inv);
            q = fminf(fmaxf(q, -128.0f), 127.0f);
            int b = (int)q;
            r |= (b & 0xFF) << (8 * j);
        }
        packed[c] = r;
    }
    v4i out = {packed[0], packed[1], packed[2], packed[3]};
    *(v4i*)(aq + i * 16) = out;
}

__global__ void pack_w_kernel(const int* __restrict__ w,
                              signed char* __restrict__ bt) {
    size_t t  = (size_t)blockIdx.x * blockDim.x + threadIdx.x;
    int    n  = (int)(t % NDIM);
    int    k0 = (int)(t / NDIM) * 8;
    int lo = 0, hi = 0;
#pragma unroll
    for (int j = 0; j < 4; ++j) {
        lo |= (w[(size_t)(k0 + j)     * NDIM + n] & 0xFF) << (8 * j);
        hi |= (w[(size_t)(k0 + 4 + j) * NDIM + n] & 0xFF) << (8 * j);
    }
    v2i out = {lo, hi};
    *(v2i*)(bt + (size_t)n * KDIM + k0) = out;
}

__global__ __launch_bounds__(256)
void gemm_w8a8_kernel(const signed char* __restrict__ aq,
                      const signed char* __restrict__ bt,
                      const float* __restrict__ bias,
                      const float* __restrict__ input_scale,
                      const float* __restrict__ weight_scale,
                      float* __restrict__ out) {
    const int lane = threadIdx.x & 31;
    const int wave = threadIdx.x >> 5;
    const int half = lane >> 4;
    const int r    = lane & 15;

    const int blockN = blockIdx.x * 128;
    const int blockM = blockIdx.y * 128;
    const int wm = blockM + (wave >> 2) * 64;
    const int wn = blockN + (wave & 3) * 32;

    const float s = input_scale[0] * weight_scale[0];

    unsigned aoff[4];
#pragma unroll
    for (int mt = 0; mt < 4; ++mt)
        aoff[mt] = (unsigned)(wm + mt * 16 + r) * KDIM + (unsigned)(half * 16);
    unsigned boff[2];
#pragma unroll
    for (int nt = 0; nt < 2; ++nt)
        boff[nt] = (unsigned)(wn + nt * 16 + r) * KDIM + (unsigned)(half * 16);

    v8i acc[4][2];
#pragma unroll
    for (int mt = 0; mt < 4; ++mt)
#pragma unroll
        for (int nt = 0; nt < 2; ++nt)
            acc[mt][nt] = (v8i){0, 0, 0, 0, 0, 0, 0, 0};

    v8i afrag[2][4];
    v8i bfrag[2][2];

    auto loadA = [&](int buf, int k0) {
        const signed char* base = aq + k0;
#pragma unroll
        for (int mt = 0; mt < 4; ++mt) {
            const signed char* p = base + (size_t)aoff[mt];
            v2i t0 = *(const v2i*)(p + 0);
            v2i t1 = *(const v2i*)(p + 8);
            v2i t2 = *(const v2i*)(p + 32);
            v2i t3 = *(const v2i*)(p + 40);
            afrag[buf][mt] = (v8i){t0.x, t0.y, t1.x, t1.y,
                                   t2.x, t2.y, t3.x, t3.y};
        }
    };
    auto loadB = [&](int buf, int k0) {
        const signed char* base = bt + k0;
#pragma unroll
        for (int nt = 0; nt < 2; ++nt) {
            const signed char* p = base + (size_t)boff[nt];
            v4i t0 = *(const v4i*)(p + 0);
            v4i t1 = *(const v4i*)(p + 32);
            bfrag[buf][nt] = (v8i){t0.x, t0.y, t0.z, t0.w,
                                   t1.x, t1.y, t1.z, t1.w};
        }
    };

    loadA(0, 0);
    loadB(0, 0);

#pragma unroll 2
    for (int k0 = 0; k0 < KDIM; k0 += 64) {
        const int cur = (k0 >> 6) & 1;
        const int nxt = cur ^ 1;

        if (k0 + 64 < KDIM) {
            loadA(nxt, k0 + 64);
            loadB(nxt, k0 + 64);
            __builtin_prefetch(aq + (size_t)aoff[0] + k0 + 128, 0, 3);
            __builtin_prefetch(bt + (size_t)boff[0] + k0 + 128, 0, 3);
        }

#pragma unroll
        for (int mt = 0; mt < 4; ++mt)
#pragma unroll
            for (int nt = 0; nt < 2; ++nt)
                acc[mt][nt] = __builtin_amdgcn_wmma_i32_16x16x64_iu8(
                     true, afrag[cur][mt],
                     true, bfrag[cur][nt],
                    acc[mt][nt],
                     false,  false);
    }

#pragma unroll
    for (int nt = 0; nt < 2; ++nt) {
        const int col = wn + nt * 16 + r;
        const float bv = bias[col];
#pragma unroll
        for (int mt = 0; mt < 4; ++mt) {
            const int m0 = wm + mt * 16 + half * 8;
            v8i c = acc[mt][nt];
#pragma unroll
            for (int i = 0; i < 8; ++i)
                out[(size_t)(m0 + i) * NDIM + col] = (float)c[i] * s + bv;
        }
    }
}

extern "C" void kernel_launch(void* const* d_in, const int* in_sizes, int n_in,
                              void* d_out, int out_size, void* d_ws, size_t ws_size,
                              hipStream_t stream) {
    const float* x            = (const float*)d_in[0];
    const int*   weight       = (const int*)  d_in[1];
    const float* bias         = (const float*)d_in[2];
    const float* input_scale  = (const float*)d_in[3];
    const float* weight_scale = (const float*)d_in[4];
    float*       out          = (float*)d_out;

    signed char* aq = (signed char*)d_ws;
    signed char* bt = aq + (size_t)MDIM * KDIM;

    {
        size_t threads = (size_t)MDIM * KDIM / 16;
        quantize_x_kernel<<<dim3((unsigned)(threads / 256)), dim3(256), 0, stream>>>(
            x, input_scale, aq);
    }
    {
        size_t threads = (size_t)NDIM * (KDIM / 8);
        pack_w_kernel<<<dim3((unsigned)(threads / 256)), dim3(256), 0, stream>>>(
            weight, bt);
    }
    {
        dim3 grid(NDIM / 128, MDIM / 128);
        gemm_w8a8_kernel<<<grid, dim3(256), 0, stream>>>(
            aq, bt, bias, input_scale, weight_scale, out);
    }
}
